// MOE_6640019440403
// MI455X (gfx1250) — matrix operand order repaired and checked by reading, not compiled, not run on hardware
//
// MoE top-2 FFN for MI455X (gfx1250), wave32 + WMMA bf16.
// Roofline: 0.55 TFLOP vs ~400MB HBM -> AI ~1375 flop/B >> ~215 machine balance
// => compute bound => v_wmma_f32_16x16x32_bf16 is the workhorse.
// Key optimization vs prev round: weights are pre-swizzled (during the one-shot
// f32->bf16 convert) into WMMA B-fragment tile order, so the hot loop fetches
// each B operand with two wave-coalesced global_load_b128 (L2-resident) and
// each A operand with two ds_load_b128 -- no per-K-step LDS staging/barriers.
#include <hip/hip_runtime.h>

#define Bdim 4
#define Tdim 4096
#define Cdim 1024
#define Edim 8
#define Kdim 2
#define Hdim 4096
#define Ntok (Bdim * Tdim)   // 16384

typedef __attribute__((ext_vector_type(16))) __bf16 v16bf;
typedef __attribute__((ext_vector_type(8)))  float  v8f;
static_assert(sizeof(v16bf) == 32, "v16bf must be 8 VGPRs");

static __device__ __forceinline__ unsigned short f2bf(float f) {
    unsigned int u = __builtin_bit_cast(unsigned int, f);
    u += 0x7fffu + ((u >> 16) & 1u);          // round-to-nearest-even
    return (unsigned short)(u >> 16);
}

// ---- WMMA operand loaders ---------------------------------------------------
// A matrix 16x32 bf16 (ISA 7.12.2): lane L holds row M=row0+L%16;
// K contiguous in lane (this code's order; B tiles match): ushorts [k0+koff .. +7] and [k0+koff+8 .. +15],
// koff = (L>=16) ? 16 : 0  ==> two 16B LDS loads.
static __device__ __forceinline__ v16bf load_a16(const unsigned short* s, int ld,
                                                 int row0, int k0) {
    const int lane = threadIdx.x & 31;
    const int m    = row0 + (lane & 15);
    const int koff = (lane & 16) ? 16 : 0;
    union { v16bf v; uint4 q[2]; } r;
    const unsigned short* p = s + m * ld + k0 + koff;
    r.q[0] = *(const uint4*)(p);
    r.q[1] = *(const uint4*)(p + 8);
    return r.v;
}

// B matrix 32x16 bf16 fragment, pre-swizzled in memory as one contiguous 1KB
// tile: lane L's 32 bytes at tile_base + L*32  ==> two global_load_b128,
// wave reads 1KB contiguous.
static __device__ __forceinline__ v16bf load_bfrag(const unsigned short* tile) {
    const int lane = threadIdx.x & 31;
    union { v16bf v; uint4 q[2]; } r;
    const uint4* p = (const uint4*)tile + lane * 2;
    r.q[0] = p[0];
    r.q[1] = p[1];
    return r.v;
}

static __device__ __forceinline__ v8f wmma_bf16(v16bf a, v16bf b, v8f c) {
    return __builtin_amdgcn_wmma_f32_16x16x32_bf16(false, a, false, b,
                                                   (short)0, c, false, false);
}

// ---- small kernels ----------------------------------------------------------
__global__ void init_small_kernel(int* counts, float* prob_sum) {
    if (threadIdx.x < Edim) { counts[threadIdx.x] = 0; prob_sum[threadIdx.x] = 0.0f; }
}

__global__ void zero_out_kernel(float* out, long n4) {
    long i = (long)blockIdx.x * blockDim.x + threadIdx.x;
    const long stride = (long)gridDim.x * blockDim.x;
    const float4 z = {0.f, 0.f, 0.f, 0.f};
    for (; i < n4; i += stride) ((float4*)out)[i] = z;
}

// Convert f32 weights -> bf16, swizzled into WMMA B-fragment tiles.
// W1 [E][C][H]: tiles (tk over C/32, tn over H/16), linear L = tn*32 + tk
//   (so phase-1 K-steps stream contiguous tiles).
// W2 [E][H][C]: tiles (tk over H/32, tn over C/16), linear L = tk*64 + tn
//   (so phase-2 column tiles of one K-step stream contiguous tiles).
// One wave converts one tile: gathers 16 strided f32, stores 1KB coalesced.
__global__ void cvt_weights_tiled_kernel(const float* __restrict__ w1,
                                         const float* __restrict__ w2,
                                         unsigned short* __restrict__ w1t,
                                         unsigned short* __restrict__ w2t) {
    const int wv = threadIdx.x >> 5, l = threadIdx.x & 31;
    const int T = blockIdx.x * 8 + wv;        // 0 .. 131071
    const int nin = l & 15;
    const int kb  = (l & 16) ? 16 : 0;
    const float* src; unsigned short* dst;
    long sbase, dbase; int ld;
    if (T < 65536) {                           // W1: 8 experts x 8192 tiles
        const int e = T >> 13, L = T & 8191;
        const int tn = L >> 5, tk = L & 31;
        ld = Hdim;
        sbase = (long)e * Cdim * Hdim + (long)(tk * 32 + kb) * Hdim + tn * 16 + nin;
        dbase = (long)e * Cdim * Hdim + (long)L * 512;
        src = w1; dst = w1t;
    } else {                                   // W2: 8 experts x 8192 tiles
        const int T2 = T - 65536;
        const int e = T2 >> 13, L = T2 & 8191;
        const int tk = L >> 6, tn = L & 63;
        ld = Cdim;
        sbase = (long)e * Hdim * Cdim + (long)(tk * 32 + kb) * Cdim + tn * 16 + nin;
        dbase = (long)e * Hdim * Cdim + (long)L * 512;
        src = w2; dst = w2t;
    }
    unsigned int q[8];
#pragma unroll
    for (int v = 0; v < 8; ++v) {
        const float lo = src[sbase + (long)(2 * v) * ld];
        const float hi = src[sbase + (long)(2 * v + 1) * ld];
        q[v] = (unsigned int)f2bf(lo) | ((unsigned int)f2bf(hi) << 16);
    }
    uint4* o = (uint4*)(dst + dbase) + l * 2;
    o[0] = make_uint4(q[0], q[1], q[2], q[3]);
    o[1] = make_uint4(q[4], q[5], q[6], q[7]);
}

// One wave32 per token: logits, softmax, top-2, expert lists, aux stats.
__global__ void router_kernel(const float* __restrict__ x,
                              const float* __restrict__ rw,
                              const float* __restrict__ rb,
                              float* __restrict__ tv,      // [N,2] topk vals (flat)
                              int* counts, float* prob_sum,
                              int* __restrict__ lists) {
    const int wave = threadIdx.x >> 5, lane = threadIdx.x & 31;
    const int t = blockIdx.x * 8 + wave;
    if (t >= Ntok) return;
    float acc[Edim];
#pragma unroll
    for (int e = 0; e < Edim; ++e) acc[e] = 0.0f;
    const float* xr = x + (long)t * Cdim;
    for (int c = lane; c < Cdim; c += 32) {
        const float xv = xr[c];
        const float* r = rw + c * Edim;   // router_w row c: 8 contiguous floats
#pragma unroll
        for (int e = 0; e < Edim; ++e) acc[e] = fmaf(xv, r[e], acc[e]);
    }
#pragma unroll
    for (int e = 0; e < Edim; ++e)
#pragma unroll
        for (int off = 16; off > 0; off >>= 1) acc[e] += __shfl_xor(acc[e], off, 32);
    if (lane == 0) {
        float l[Edim], p[Edim], mx = -1e30f, s = 0.0f;
#pragma unroll
        for (int e = 0; e < Edim; ++e) { l[e] = acc[e] + rb[e]; mx = fmaxf(mx, l[e]); }
#pragma unroll
        for (int e = 0; e < Edim; ++e) { p[e] = __expf(l[e] - mx); s += p[e]; }
        const float inv = 1.0f / s;
        int i0 = 0, i1 = 0; float v0 = -1.0f, v1 = -1.0f;
#pragma unroll
        for (int e = 0; e < Edim; ++e) {
            p[e] *= inv;
            atomicAdd(&prob_sum[e], p[e] * (1.0f / (float)Ntok));  // importance
            if (p[e] > v0) { v1 = v0; i1 = i0; v0 = p[e]; i0 = e; }
            else if (p[e] > v1) { v1 = p[e]; i1 = e; }
        }
        tv[2 * t + 0] = v0;
        tv[2 * t + 1] = v1;
        int pos0 = atomicAdd(&counts[i0], 1); lists[i0 * Ntok + pos0] = t;
        int pos1 = atomicAdd(&counts[i1], 1); lists[i1 * Ntok + pos1] = t;
    }
}

__global__ void aux_kernel(const float* prob_sum, const int* counts, float* out_aux) {
    if (threadIdx.x == 0) {
        float a = 0.0f;
        for (int e = 0; e < Edim; ++e)
            a += prob_sum[e] * ((float)counts[e] / (float)(Ntok * Kdim));
        out_aux[0] = a;
    }
}

// ---- fused expert FFN: out += tv[t] * (relu(x@w1+b1) @ w2 + b2) -------------
// Block: 256 thr (8 waves), 32 gathered tokens, full C=1024 output columns in
// registers (16 v8f accumulators/lane). H processed in two 2048-col halves:
//   phase1: h[32,2048] = relu(x@W1half + b1) -> LDS (bf16)
//   phase2: acc[32,1024] += h @ W2half
// Weights arrive as pre-swizzled B-fragment tiles -> global_load_b128 streams.
// Dynamic LDS 192KB: xs 64KB | hs 128KB. Only 6 barriers per block.
__launch_bounds__(256, 1)
__global__ void moe_ffn_kernel(const unsigned short* __restrict__ w1t,
                               const unsigned short* __restrict__ w2t,
                               const float* __restrict__ b1,
                               const float* __restrict__ b2,
                               const float* __restrict__ x,
                               const float* __restrict__ tv,
                               const int* __restrict__ counts,
                               const int* __restrict__ lists,
                               float* __restrict__ out) {
    extern __shared__ char smem[];
    unsigned short* xs = (unsigned short*)smem;              // [32][1024]
    unsigned short* hs = (unsigned short*)(smem + 65536);    // [32][2048]
    __shared__ int   toklds[32];
    __shared__ float sclds[32];

    const int e   = blockIdx.y;
    const int cnt = counts[e];
    const int m0  = blockIdx.x * 32;
    if (m0 >= cnt) return;

    const int tid  = threadIdx.x;
    const int wv   = tid >> 5;
    const int lane = tid & 31;

    if (tid < 32) {
        const int g = m0 + tid;
        int tok = 0; float sc = 0.0f;
        if (g < cnt) { tok = lists[e * Ntok + g]; sc = tv[tok]; }
        toklds[tid] = tok; sclds[tid] = sc;   // invalid rows: scale 0 -> add 0
    }
    __syncthreads();

    // Gather + convert x rows to bf16 in LDS (thread: 128 contiguous cols).
    {
        const int r  = tid >> 3;
        const int c0 = (tid & 7) * 128;
        const float* src = x + (long)toklds[r] * Cdim + c0;
        unsigned short* dst = xs + r * Cdim + c0;
#pragma unroll
        for (int i = 0; i < 128; i += 4) {
            float4 f = *(const float4*)(src + i);
            dst[i + 0] = f2bf(f.x); dst[i + 1] = f2bf(f.y);
            dst[i + 2] = f2bf(f.z); dst[i + 3] = f2bf(f.w);
        }
    }
    __syncthreads();

    const unsigned short* w1e = w1t + (long)e * Cdim * Hdim;
    const unsigned short* w2e = w2t + (long)e * Hdim * Cdim;

    v8f acc[16];
#pragma unroll
    for (int i = 0; i < 16; ++i) acc[i] = (v8f){};

    for (int hc = 0; hc < 2; ++hc) {
        const int hbase = hc * 2048;
        // -------- phase 1: h chunk [32 x 2048], no barriers ---------------
        for (int g = 0; g < 16; ++g) {          // 128 cols per group (16/wave)
            const int tn = hc * 128 + g * 8 + wv;       // column tile index
            const unsigned short* bt = w1e + (long)(tn * 32) * 512;
            v8f h0 = (v8f){}, h1 = (v8f){};
#pragma unroll 4
            for (int kk = 0; kk < 32; ++kk) {   // reduce over C
                v16bf bb = load_bfrag(bt + (long)kk * 512);
                if (kk < 31)
                    __builtin_prefetch((const char*)(bt + (long)(kk + 1) * 512), 0, 1);
                v16bf a0 = load_a16(xs, Cdim, 0,  kk * 32);
                v16bf a1 = load_a16(xs, Cdim, 16, kk * 32);
                h0 = wmma_bf16(a0, bb, h0);
                h1 = wmma_bf16(a1, bb, h1);
            }
            // epilogue: +b1, relu, bf16 -> hs
            const int coln = g * 128 + wv * 16 + (lane & 15);
            const float bias = b1[e * Hdim + hbase + coln];
#pragma unroll
            for (int r = 0; r < 8; ++r) {
                const int m = r + ((lane & 16) ? 8 : 0);
                hs[m * 2048 + coln]        = f2bf(fmaxf(h0[r] + bias, 0.0f));
                hs[(m + 16) * 2048 + coln] = f2bf(fmaxf(h1[r] + bias, 0.0f));
            }
        }
        __syncthreads();   // hs complete before any wave reads other cols
        // -------- phase 2: acc += h @ W2chunk, no staging -----------------
        for (int kk = 0; kk < 64; ++kk) {       // reduce over 2048 chunk rows
            const int tk = hc * 64 + kk;
            v16bf a0 = load_a16(hs, 2048, 0,  kk * 32);
            v16bf a1 = load_a16(hs, 2048, 16, kk * 32);
            const unsigned short* bt = w2e + (long)(tk * 64 + wv * 8) * 512;
            if (kk < 63)
                __builtin_prefetch((const char*)(bt + (long)64 * 512), 0, 1);
#pragma unroll
            for (int j = 0; j < 8; ++j) {       // wave owns cols [wv*128, +128)
                v16bf bb = load_bfrag(bt + (long)j * 512);
                acc[2 * j]     = wmma_bf16(a0, bb, acc[2 * j]);
                acc[2 * j + 1] = wmma_bf16(a1, bb, acc[2 * j + 1]);
            }
        }
        __syncthreads();   // phase-2 reads done before next chunk rewrites hs
    }
    // -------- scatter: out[tok] += scale * (acc + b2) ----------------------
#pragma unroll
    for (int j = 0; j < 8; ++j) {
        const int col = wv * 128 + j * 16 + (lane & 15);
        const float bias2 = b2[e * Cdim + col];
#pragma unroll
        for (int r = 0; r < 8; ++r) {
            const int m = r + ((lane & 16) ? 8 : 0);
            const float v0 = (acc[2 * j][r] + bias2) * sclds[m];
            const float v1 = (acc[2 * j + 1][r] + bias2) * sclds[m + 16];
            unsafeAtomicAdd(&out[(long)toklds[m] * Cdim + col], v0);
            unsafeAtomicAdd(&out[(long)toklds[m + 16] * Cdim + col], v1);
        }
    }
}

// ---- host -------------------------------------------------------------------
extern "C" void kernel_launch(void* const* d_in, const int* in_sizes, int n_in,
                              void* d_out, int out_size, void* d_ws, size_t ws_size,
                              hipStream_t stream) {
    const float* x  = (const float*)d_in[0];
    const float* rw = (const float*)d_in[1];
    const float* rb = (const float*)d_in[2];
    const float* w1 = (const float*)d_in[3];
    const float* b1 = (const float*)d_in[4];
    const float* w2 = (const float*)d_in[5];
    const float* b2 = (const float*)d_in[6];
    float* out = (float*)d_out;

    // workspace layout (needs ~135MB)
    char* ws = (char*)d_ws;
    unsigned short* w1t = (unsigned short*)ws;                       // 64 MiB
    unsigned short* w2t = (unsigned short*)(ws + 67108864);          // 64 MiB
    float* tv     = (float*)(ws + 134217728);                        // 128 KiB
    int*   counts = (int*)  (ws + 134217728 + 131072);
    float* psum   = (float*)(ws + 134217728 + 131072 + 256);
    int*   lists  = (int*)  (ws + 134217728 + 131072 + 512);         // 512 KiB

    (void)in_sizes; (void)n_in; (void)out_size; (void)ws_size;

    hipFuncSetAttribute((const void*)moe_ffn_kernel,
                        hipFuncAttributeMaxDynamicSharedMemorySize, 196608);

    init_small_kernel<<<1, 32, 0, stream>>>(counts, psum);
    zero_out_kernel<<<2048, 256, 0, stream>>>(out, (long)Ntok * Cdim / 4);
    cvt_weights_tiled_kernel<<<16384, 256, 0, stream>>>(w1, w2, w1t, w2t);
    router_kernel<<<Ntok / 8, 256, 0, stream>>>(x, rw, rb, tv, counts, psum, lists);
    moe_ffn_kernel<<<dim3(Ntok / 32, Edim), 256, 196608, stream>>>(
        w1t, w2t, b1, b2, x, tv, counts, lists, out);
    aux_kernel<<<1, 32, 0, stream>>>(psum, counts, out + (long)Ntok * Cdim);
}
